// Model_35708358099201
// MI455X (gfx1250) — matrix operand order repaired and checked by reading, not compiled, not run on hardware
//
#include <hip/hip_runtime.h>
#include <hip/hip_bf16.h>

typedef _Float16 v16h __attribute__((ext_vector_type(16)));
typedef float    v8f  __attribute__((ext_vector_type(8)));

#define GN   10000
#define DN   10000
#define EXN  320000
#define EYN  320000
#define FDIM 128

// ---------------------------------------------------------------- utilities
__global__ void k_fill(float* __restrict__ p, float v, int n) {
    int i = blockIdx.x * blockDim.x + threadIdx.x;
    if (i < n) p[i] = v;
}

// ew[e] = dense[row[e], col[e]];  deg[col[e]] += ew[e]
__global__ void k_edge_w_deg(const float* __restrict__ dense,
                             const int* __restrict__ row, const int* __restrict__ col,
                             float* __restrict__ ew, float* __restrict__ deg,
                             int E, int N) {
    int e = blockIdx.x * blockDim.x + threadIdx.x;
    if (e >= E) return;
    int r = row[e], c = col[e];
    float w = dense[(size_t)r * (size_t)N + (size_t)c];
    ew[e] = w;
    unsafeAtomicAdd(&deg[c], w);
}

__global__ void k_rsqrt_inplace(float* __restrict__ p, int n) {
    int i = blockIdx.x * blockDim.x + threadIdx.x;
    if (i < n) p[i] = rsqrtf(p[i]);
}

__global__ void k_norm(const float* __restrict__ ew,
                       const int* __restrict__ row, const int* __restrict__ col,
                       const float* __restrict__ dinv, float* __restrict__ norm, int E) {
    int e = blockIdx.x * blockDim.x + threadIdx.x;
    if (e >= E) return;
    norm[e] = dinv[row[e]] * ew[e] * dinv[col[e]];
}

// agg[col[e], f..f+3] += H[row[e], f..f+3] * norm[e]   (4 features per thread)
__global__ void k_scatter(const float* __restrict__ H,
                          const int* __restrict__ row, const int* __restrict__ col,
                          const float* __restrict__ norm, float* __restrict__ agg,
                          int E) {
    long long idx = (long long)blockIdx.x * blockDim.x + threadIdx.x;
    if (idx >= (long long)E * (FDIM / 4)) return;
    int e = (int)(idx >> 5);                 // FDIM/4 == 32 quads per edge
    int f = ((int)(idx & 31)) << 2;
    float w = norm[e];
    const float4 h = *(const float4*)(H + (size_t)row[e] * FDIM + f);
    float* dst = agg + (size_t)col[e] * FDIM + f;
    unsafeAtomicAdd(dst + 0, h.x * w);
    unsafeAtomicAdd(dst + 1, h.y * w);
    unsafeAtomicAdd(dst + 2, h.z * w);
    unsafeAtomicAdd(dst + 3, h.w * w);
}

// agg = relu(agg + H * dinv^2 + bias)   (self-loop contribution + bias + relu)
__global__ void k_selfloop_bias_relu(float* __restrict__ agg, const float* __restrict__ H,
                                     const float* __restrict__ dinv,
                                     const float* __restrict__ bias, int Nnodes) {
    int idx = blockIdx.x * blockDim.x + threadIdx.x;
    if (idx >= Nnodes * FDIM) return;
    int n = idx >> 7;
    int f = idx & (FDIM - 1);
    float d = dinv[n];
    float v = agg[idx] + H[idx] * d * d + bias[f];
    agg[idx] = fmaxf(v, 0.0f);
}

// --------------------------------------------------------- fragment loaders
// A fragment (16x32 f16) for one 16-row tile, fp32 source, row-major, lda = K.
// lanes<16 hold K = k0+0..15 ; lanes>=16: +16.  Requires kb 32B-aligned.
__device__ __forceinline__ v16h load_a_frag(const float* __restrict__ ap, int kb) {
    const float4 f0 = *(const float4*)(ap + kb);
    const float4 f1 = *(const float4*)(ap + kb + 4);
    const float4 f2 = *(const float4*)(ap + kb + 8);
    const float4 f3 = *(const float4*)(ap + kb + 12);
    v16h a;
    a[0]  = (_Float16)f0.x; a[1]  = (_Float16)f0.y; a[2]  = (_Float16)f0.z; a[3]  = (_Float16)f0.w;
    a[4]  = (_Float16)f1.x; a[5]  = (_Float16)f1.y; a[6]  = (_Float16)f1.z; a[7]  = (_Float16)f1.w;
    a[8]  = (_Float16)f2.x; a[9]  = (_Float16)f2.y; a[10] = (_Float16)f2.z; a[11] = (_Float16)f2.w;
    a[12] = (_Float16)f3.x; a[13] = (_Float16)f3.y; a[14] = (_Float16)f3.z; a[15] = (_Float16)f3.w;
    return a;
}

// ------------------------------------------------- WMMA GEMM: C = A @ W (+b, relu)
// A: [M x K] fp32 row-major, W: [K x N] fp32 row-major, C: [M x N] fp32.
// Block = 256 threads = 8 waves.  Wave computes a 16 x N strip (N = 16*NT).
// A fragment loaded once per k-step and reused across all NT n-tiles.
// Requires M % 16 == 0, K % 32 == 0.
template <int NT>
__global__ void k_gemm_f16(const float* __restrict__ A, const float* __restrict__ W,
                           const float* __restrict__ bias, float* __restrict__ C,
                           int M, int N, int K, int fuse_relu) {
    const int lane = threadIdx.x & 31;
    const int wave = threadIdx.x >> 5;
    const int hi   = (lane & 16) ? 1 : 0;
    const int mt   = blockIdx.x * 8 + wave;          // 16-row tile index
    if (mt * 16 >= M) return;                        // uniform per wave

    const int l15  = lane & 15;
    const int mrow = mt * 16 + l15;
    const float* ap = A + (size_t)mrow * (size_t)K;

    v8f acc[NT];
#pragma unroll
    for (int t = 0; t < NT; ++t) acc[t] = v8f{};

    for (int k0 = 0; k0 < K; k0 += 32) {
        const v16h a = load_a_frag(ap, k0 + (hi ? 16 : 0));
        const int kb = k0 + (hi ? 16 : 0);
#pragma unroll
        for (int t = 0; t < NT; ++t) {
            const int ncol = t * 16 + l15;
            v16h b;
#pragma unroll
            for (int i = 0; i < 16; ++i)
                b[i] = (_Float16)W[(size_t)(kb + i) * (size_t)N + (size_t)ncol];
            acc[t] = __builtin_amdgcn_wmma_f32_16x16x32_f16(false, a, false, b,
                                                            (short)0, acc[t], false, false);
        }
    }
#pragma unroll
    for (int t = 0; t < NT; ++t) {
        const int ncol = t * 16 + l15;
        const float bv = bias ? bias[ncol] : 0.0f;
#pragma unroll
        for (int v = 0; v < 8; ++v) {
            const int r = mt * 16 + v + (hi ? 8 : 0);
            float val = acc[t][v] + bv;
            if (fuse_relu) val = fmaxf(val, 0.0f);
            C[(size_t)r * (size_t)N + (size_t)ncol] = val;
        }
    }
}

// ------------------------------------------------- final GEMM: C = A @ B^T, K = 32
// A: [M x 32], B: [Nn x 32], C: [M x Nn].  Block = 256 threads = 8 waves.
// Each wave keeps one A fragment (single WMMA covers K=32) and sweeps 8 N-tiles.
// Requires M % 16 == 0, Nn % 16 == 0.
__global__ void k_final_gemm(const float* __restrict__ A, const float* __restrict__ B,
                             float* __restrict__ C, int M, int Nn) {
    const int lane = threadIdx.x & 31;
    const int wave = threadIdx.x >> 5;
    const int hi   = (lane & 16) ? 1 : 0;
    const int l15  = lane & 15;
    const int mt   = blockIdx.y * 8 + wave;          // 16-row tile index
    if (mt * 16 >= M) return;                        // uniform per wave

    const v16h a = load_a_frag(A + (size_t)(mt * 16 + l15) * 32, hi ? 16 : 0);

    const int nb0 = blockIdx.x * 128;
    const int kb  = hi ? 16 : 0;
    for (int nt = 0; nt < 8; ++nt) {
        const int n_base = nb0 + nt * 16;
        if (n_base >= Nn) break;                     // uniform per wave
        const int n = n_base + l15;
        const float* bp = B + (size_t)n * 32;        // B^T[k][n] = B[n][k]
        const float4 f0 = *(const float4*)(bp + kb);
        const float4 f1 = *(const float4*)(bp + kb + 4);
        const float4 f2 = *(const float4*)(bp + kb + 8);
        const float4 f3 = *(const float4*)(bp + kb + 12);
        v16h b;
        b[0]  = (_Float16)f0.x; b[1]  = (_Float16)f0.y; b[2]  = (_Float16)f0.z; b[3]  = (_Float16)f0.w;
        b[4]  = (_Float16)f1.x; b[5]  = (_Float16)f1.y; b[6]  = (_Float16)f1.z; b[7]  = (_Float16)f1.w;
        b[8]  = (_Float16)f2.x; b[9]  = (_Float16)f2.y; b[10] = (_Float16)f2.z; b[11] = (_Float16)f2.w;
        b[12] = (_Float16)f3.x; b[13] = (_Float16)f3.y; b[14] = (_Float16)f3.z; b[15] = (_Float16)f3.w;

        v8f acc = {};
        acc = __builtin_amdgcn_wmma_f32_16x16x32_f16(false, a, false, b,
                                                     (short)0, acc, false, false);
#pragma unroll
        for (int v = 0; v < 8; ++v) {
            const int r = mt * 16 + v + (hi ? 8 : 0);
            C[(size_t)r * (size_t)Nn + (size_t)n] = acc[v];
        }
    }
}

// ---------------------------------------------------------------- host side
static inline int cdiv(long long a, long long b) { return (int)((a + b - 1) / b); }

struct GraphWs {
    float *ew, *norm, *dinv, *H, *A, *B, *M64, *M32;
};

static void run_gcn_conv(const float* X, const float* W, const float* b,
                         const int* row, const int* col,
                         const GraphWs& g, float* outbuf, int Nnodes, int E,
                         hipStream_t stream) {
    k_gemm_f16<8><<<cdiv(Nnodes, 128), 256, 0, stream>>>(X, W, nullptr, g.H,
                                                         Nnodes, FDIM, FDIM, 0);
    k_fill<<<cdiv((long long)Nnodes * FDIM, 256), 256, 0, stream>>>(outbuf, 0.0f, Nnodes * FDIM);
    k_scatter<<<cdiv((long long)E * (FDIM / 4), 256), 256, 0, stream>>>(g.H, row, col, g.norm, outbuf, E);
    k_selfloop_bias_relu<<<cdiv((long long)Nnodes * FDIM, 256), 256, 0, stream>>>(outbuf, g.H, g.dinv, b, Nnodes);
}

extern "C" void kernel_launch(void* const* d_in, const int* in_sizes, int n_in,
                              void* d_out, int out_size, void* d_ws, size_t ws_size,
                              hipStream_t stream) {
    const float* x_g    = (const float*)d_in[0];
    const float* x_d    = (const float*)d_in[1];
    const float* data_x = (const float*)d_in[2];
    const float* data_y = (const float*)d_in[3];
    const int*   ei_x   = (const int*)d_in[4];
    const int*   ei_y   = (const int*)d_in[5];
    const float* Wx1 = (const float*)d_in[6],  *bx1 = (const float*)d_in[7];
    const float* Wx2 = (const float*)d_in[8],  *bx2 = (const float*)d_in[9];
    const float* Wy1 = (const float*)d_in[10], *by1 = (const float*)d_in[11];
    const float* Wy2 = (const float*)d_in[12], *by2 = (const float*)d_in[13];
    const float* lx1W = (const float*)d_in[14], *lx1b = (const float*)d_in[15];
    const float* lx2W = (const float*)d_in[16], *lx2b = (const float*)d_in[17];
    const float* lx3W = (const float*)d_in[18], *lx3b = (const float*)d_in[19];
    const float* ly1W = (const float*)d_in[20], *ly1b = (const float*)d_in[21];
    const float* ly2W = (const float*)d_in[22], *ly2b = (const float*)d_in[23];
    const float* ly3W = (const float*)d_in[24], *ly3b = (const float*)d_in[25];
    float* out = (float*)d_out;

    float* ws = (float*)d_ws;
    size_t off = 0;
    auto alloc = [&](size_t n) { float* p = ws + off; off += n; return p; };

    GraphWs gx, gy;
    gx.ew  = alloc(EXN);            gx.norm = alloc(EXN);
    gx.dinv = alloc(GN);
    gx.H   = alloc((size_t)GN * FDIM);
    gx.A   = alloc((size_t)GN * FDIM);
    gx.B   = alloc((size_t)GN * FDIM);
    gx.M64 = alloc((size_t)GN * 64);
    gx.M32 = alloc((size_t)GN * 32);
    gy.ew  = alloc(EYN);            gy.norm = alloc(EYN);
    gy.dinv = alloc(DN);
    gy.H   = alloc((size_t)DN * FDIM);
    gy.A   = alloc((size_t)DN * FDIM);
    gy.B   = alloc((size_t)DN * FDIM);
    gy.M64 = alloc((size_t)DN * 64);
    gy.M32 = alloc((size_t)DN * 32);

    const int* rx = ei_x;        const int* cx = ei_x + EXN;
    const int* ry = ei_y;        const int* cy = ei_y + EYN;

    // ---- graph normalization terms
    k_fill<<<cdiv(GN, 256), 256, 0, stream>>>(gx.dinv, 1.0f, GN);   // deg (incl self-loop)
    k_edge_w_deg<<<cdiv(EXN, 256), 256, 0, stream>>>(data_x, rx, cx, gx.ew, gx.dinv, EXN, GN);
    k_rsqrt_inplace<<<cdiv(GN, 256), 256, 0, stream>>>(gx.dinv, GN);
    k_norm<<<cdiv(EXN, 256), 256, 0, stream>>>(gx.ew, rx, cx, gx.dinv, gx.norm, EXN);

    k_fill<<<cdiv(DN, 256), 256, 0, stream>>>(gy.dinv, 1.0f, DN);
    k_edge_w_deg<<<cdiv(EYN, 256), 256, 0, stream>>>(data_y, ry, cy, gy.ew, gy.dinv, EYN, DN);
    k_rsqrt_inplace<<<cdiv(DN, 256), 256, 0, stream>>>(gy.dinv, DN);
    k_norm<<<cdiv(EYN, 256), 256, 0, stream>>>(gy.ew, ry, cy, gy.dinv, gy.norm, EYN);

    // ---- GCN stacks (conv -> relu, x3; layers 2 and 3 share weights)
    run_gcn_conv(x_g,  Wx1, bx1, rx, cx, gx, gx.A, GN, EXN, stream);   // X1
    run_gcn_conv(gx.A, Wx2, bx2, rx, cx, gx, gx.B, GN, EXN, stream);   // X
    run_gcn_conv(gx.B, Wx2, bx2, rx, cx, gx, gx.A, GN, EXN, stream);   // X2

    run_gcn_conv(x_d,  Wy1, by1, ry, cy, gy, gy.A, DN, EYN, stream);   // Y1
    run_gcn_conv(gy.A, Wy2, by2, ry, cy, gy, gy.B, DN, EYN, stream);   // Y
    run_gcn_conv(gy.B, Wy2, by2, ry, cy, gy, gy.A, DN, EYN, stream);   // Y2

    // ---- MLPs (bias + relu fused into the WMMA GEMM epilogue)
    k_gemm_f16<8><<<cdiv(GN, 128), 256, 0, stream>>>(gx.A,   lx1W, lx1b, gx.B,   GN, 128, 128, 1);
    k_gemm_f16<4><<<cdiv(GN, 128), 256, 0, stream>>>(gx.B,   lx2W, lx2b, gx.M64, GN, 64,  128, 1);
    k_gemm_f16<2><<<cdiv(GN, 128), 256, 0, stream>>>(gx.M64, lx3W, lx3b, gx.M32, GN, 32,  64,  1);

    k_gemm_f16<8><<<cdiv(DN, 128), 256, 0, stream>>>(gy.A,   ly1W, ly1b, gy.B,   DN, 128, 128, 1);
    k_gemm_f16<4><<<cdiv(DN, 128), 256, 0, stream>>>(gy.B,   ly2W, ly2b, gy.M64, DN, 64,  128, 1);
    k_gemm_f16<2><<<cdiv(DN, 128), 256, 0, stream>>>(gy.M64, ly3W, ly3b, gy.M32, DN, 32,  64,  1);

    // ---- final: out[G, D] = x32 @ y32^T   (the 400 MB store dominates runtime)
    dim3 fg((unsigned)cdiv(DN, 128), (unsigned)cdiv(GN, 128));
    k_final_gemm<<<fg, 256, 0, stream>>>(gx.M32, gy.M32, out, GN, DN);
}
